// MultiModelWithScalarHeads_18932215841412
// MI455X (gfx1250) — matrix operand order repaired and checked by reading, not compiled, not run on hardware
//
#include <hip/hip_runtime.h>

// ---------------------------------------------------------------------------
// MultiModelWithScalarHeads — fused 3-layer head for gfx1250 (MI455X, wave32)
//
// M=2 models, rows = B*S = 8192, H=4096 -> HID=256 -> IDIM=128 -> 1 score.
// Memory-bound on streaming h (268 MB @ 23.3 TB/s ~= 12 us); everything else
// (weights, x1, x2) lives in L2 / LDS. Matrix math via v_wmma_f32_16x16x32_bf16.
// ---------------------------------------------------------------------------

typedef __bf16 bf16;
typedef __attribute__((ext_vector_type(16))) __bf16 v16bf;
typedef __attribute__((ext_vector_type(8)))  __bf16 v8bf;
typedef __attribute__((ext_vector_type(8)))  float  v8f;
typedef __attribute__((ext_vector_type(4)))  float  f32x4;

#define NMODEL 2
#define HDIM   4096
#define HID    256
#define IDIM   128
#define NROWS  8192    // B*S per model
#define TILE_R 128     // rows per workgroup

// ---------------------------------------------------------------------------
// Prep: fold BN (eval) + linear bias into per-feature scale/bias, and convert
// w1 / w2 to bf16 in workspace.
//   y = (Wx + b)*inv + (beta - mean*inv),  inv = gamma*rsqrt(var+eps)
//     = (Wx)*inv + [(b - mean)*inv + beta]
// ---------------------------------------------------------------------------
__global__ void prep_kernel(const float* __restrict__ w1, const float* __restrict__ b1,
                            const float* __restrict__ g1, const float* __restrict__ be1,
                            const float* __restrict__ m1, const float* __restrict__ v1,
                            const float* __restrict__ w2, const float* __restrict__ b2,
                            const float* __restrict__ g2, const float* __restrict__ be2,
                            const float* __restrict__ m2, const float* __restrict__ v2,
                            bf16* __restrict__ w1bf, bf16* __restrict__ w2bf,
                            float* __restrict__ scale1, float* __restrict__ bias1,
                            float* __restrict__ scale2, float* __restrict__ bias2)
{
    const int i = blockIdx.x * blockDim.x + threadIdx.x;
    const int nthreads = gridDim.x * blockDim.x;
    for (int idx = i; idx < NMODEL * HID * HDIM; idx += nthreads)
        w1bf[idx] = (bf16)w1[idx];
    if (i < NMODEL * IDIM * HID)
        w2bf[i] = (bf16)w2[i];
    if (i < NMODEL * HID) {
        float inv = g1[i] * rsqrtf(v1[i] + 1e-5f);
        scale1[i] = inv;
        bias1[i]  = (b1[i] - m1[i]) * inv + be1[i];
    }
    if (i < NMODEL * IDIM) {
        float inv = g2[i] * rsqrtf(v2[i] + 1e-5f);
        scale2[i] = inv;
        bias2[i]  = (b2[i] - m2[i]) * inv + be2[i];
    }
}

// ---------------------------------------------------------------------------
// Fused head. Grid: (NROWS/TILE_R, NMODEL). Block: 256 threads = 8 waves.
// Wave w owns rows [w*16, w*16+16) of the 128-row tile, all 256 HID columns.
//
// bf16 WMMA fragments as used here (wave32, 16x16x32; A and B share a contiguous k order):
//   A (16x32): lane l -> row = l&15; hi = l>>4; elem  e  : K = 16*hi + e
//   B (32x16): lane l -> col = l&15; hi = l>>4; elem  e  : K = 16*hi + e
//   C/D (16x16 f32): lane l -> col = l&15; vgpr v -> row = v + 8*(l>>4)
// ---------------------------------------------------------------------------
__launch_bounds__(256)
__global__ void fused_head_kernel(const float* __restrict__ h,
                                  const bf16* __restrict__ w1bf,
                                  const bf16* __restrict__ w2bf,
                                  const float* __restrict__ scale1, const float* __restrict__ bias1,
                                  const float* __restrict__ scale2, const float* __restrict__ bias2,
                                  const float* __restrict__ w3, const float* __restrict__ b3,
                                  float* __restrict__ out)
{
    __shared__ __attribute__((aligned(16))) bf16  x1[TILE_R * HID];   // 64 KB
    __shared__ __attribute__((aligned(16))) float x2[TILE_R * IDIM];  // 64 KB

    const int m    = blockIdx.y;
    const int row0 = blockIdx.x * TILE_R;
    const int tid  = threadIdx.x;
    const int lane = tid & 31;
    const int wave = tid >> 5;          // 0..7
    const int r    = lane & 15;
    const int hi   = lane >> 4;

    // ---------------- GEMM1: x1 = relu(bn1(h @ w1^T + b1)) ----------------
    // A: h rows (fp32 -> bf16 in regs).  B: w1bf rows (contiguous 32B loads).
    const float* aPtr = h + ((size_t)m * NROWS + row0 + wave * 16 + r) * HDIM + 16 * hi;
    const bf16*  w1b  = w1bf + (size_t)m * HID * HDIM + (size_t)r * HDIM + 16 * hi;

    v8f acc[16];
    #pragma unroll
    for (int t = 0; t < 16; ++t) acc[t] = {};

    for (int k0 = 0; k0 < HDIM; k0 += 32) {
        f32x4 a0 = *(const f32x4*)(aPtr + k0);
        f32x4 a1 = *(const f32x4*)(aPtr + k0 + 4);
        f32x4 a2 = *(const f32x4*)(aPtr + k0 + 8);
        f32x4 a3 = *(const f32x4*)(aPtr + k0 + 12);
        v16bf af;
        #pragma unroll
        for (int e = 0; e < 4; ++e) {
            af[e]      = (bf16)a0[e];
            af[e + 4]  = (bf16)a1[e];
            af[e + 8]  = (bf16)a2[e];
            af[e + 12] = (bf16)a3[e];
        }
        #pragma unroll
        for (int t = 0; t < 16; ++t) {
            v16bf bfr = *(const v16bf*)(w1b + (size_t)(t * 16) * HDIM + k0);
            acc[t] = __builtin_amdgcn_wmma_f32_16x16x32_bf16(
                false, af, false, bfr, (short)0, acc[t], false, false);
        }
    }

    // BN1 + ReLU -> bf16 x1 in LDS
    #pragma unroll
    for (int t = 0; t < 16; ++t) {
        const int n = t * 16 + r;
        const float sc = scale1[m * HID + n];
        const float bi = bias1[m * HID + n];
        #pragma unroll
        for (int v = 0; v < 8; ++v) {
            const int row = wave * 16 + v + 8 * hi;
            float y = acc[t][v] * sc + bi;
            y = y > 0.0f ? y : 0.0f;
            x1[row * HID + n] = (bf16)y;
        }
    }
    __syncthreads();

    // ---------------- GEMM2: x2 = relu(bn2(x1 @ w2^T + b2)) ----------------
    const bf16* aL  = &x1[(wave * 16 + r) * HID + 16 * hi];
    const bf16* w2b = w2bf + (size_t)m * IDIM * HID + (size_t)r * HID + 16 * hi;

    v8f acc2[8];
    #pragma unroll
    for (int t = 0; t < 8; ++t) acc2[t] = {};

    #pragma unroll
    for (int k0 = 0; k0 < HID; k0 += 32) {
        v8bf lo = *(const v8bf*)(aL + k0);        // K = k0 + 16*hi + 0..7
        v8bf hh = *(const v8bf*)(aL + k0 + 8);    // K = k0 + 16*hi + 8..15
        v16bf af;
        #pragma unroll
        for (int e = 0; e < 8; ++e) { af[e] = lo[e]; af[e + 8] = hh[e]; }
        #pragma unroll
        for (int t = 0; t < 8; ++t) {
            v16bf bfr = *(const v16bf*)(w2b + (size_t)(t * 16) * HID + k0);
            acc2[t] = __builtin_amdgcn_wmma_f32_16x16x32_bf16(
                false, af, false, bfr, (short)0, acc2[t], false, false);
        }
    }

    // BN2 + ReLU -> fp32 x2 in LDS
    #pragma unroll
    for (int t = 0; t < 8; ++t) {
        const int n = t * 16 + r;
        const float sc = scale2[m * IDIM + n];
        const float bi = bias2[m * IDIM + n];
        #pragma unroll
        for (int v = 0; v < 8; ++v) {
            const int row = wave * 16 + v + 8 * hi;
            float y = acc2[t][v] * sc + bi;
            y = y > 0.0f ? y : 0.0f;
            x2[row * IDIM + n] = y;
        }
    }
    __syncthreads();

    // ---------------- Final: score = x2 . w3 + b3 -> out[b,s,m] ----------------
    if (tid < TILE_R) {
        const float* w3p = w3 + m * IDIM;
        const float* xr  = &x2[tid * IDIM];
        float s = b3[m];
        #pragma unroll 8
        for (int j = 0; j < IDIM; ++j) s += xr[j] * w3p[j];
        out[(size_t)(row0 + tid) * NMODEL + m] = s;
    }
}

// ---------------------------------------------------------------------------
extern "C" void kernel_launch(void* const* d_in, const int* in_sizes, int n_in,
                              void* d_out, int out_size, void* d_ws, size_t ws_size,
                              hipStream_t stream) {
    const float* h   = (const float*)d_in[0];
    const float* w1  = (const float*)d_in[1];
    const float* b1  = (const float*)d_in[2];
    const float* g1  = (const float*)d_in[3];
    const float* be1 = (const float*)d_in[4];
    const float* m1  = (const float*)d_in[5];
    const float* v1  = (const float*)d_in[6];
    const float* w2  = (const float*)d_in[7];
    const float* b2  = (const float*)d_in[8];
    const float* g2  = (const float*)d_in[9];
    const float* be2 = (const float*)d_in[10];
    const float* m2  = (const float*)d_in[11];
    const float* v2  = (const float*)d_in[12];
    const float* w3  = (const float*)d_in[13];
    const float* b3  = (const float*)d_in[14];
    float* out = (float*)d_out;

    // Workspace carve-up (~4.33 MB total)
    char* ws = (char*)d_ws;
    bf16* w1bf = (bf16*)ws;                                     // 2*256*4096*2 = 4,194,304 B
    bf16* w2bf = (bf16*)(ws + (size_t)NMODEL * HID * HDIM * 2); // 2*128*256*2  =   131,072 B
    float* scale1 = (float*)((char*)w2bf + (size_t)NMODEL * IDIM * HID * 2);
    float* bias1  = scale1 + NMODEL * HID;
    float* scale2 = bias1  + NMODEL * HID;
    float* bias2  = scale2 + NMODEL * IDIM;

    prep_kernel<<<256, 256, 0, stream>>>(w1, b1, g1, be1, m1, v1,
                                         w2, b2, g2, be2, m2, v2,
                                         w1bf, w2bf, scale1, bias1, scale2, bias2);

    dim3 grid(NROWS / TILE_R, NMODEL);   // 64 x 2
    fused_head_kernel<<<grid, 256, 0, stream>>>(h, w1bf, w2bf,
                                                scale1, bias1, scale2, bias2,
                                                w3, b3, out);
}
